// MFNClassifier_90168543412258
// MI455X (gfx1250) — matrix operand order repaired and checked by reading, not compiled, not run on hardware
//
#include <hip/hip_runtime.h>
#include <hip/hip_bf16.h>

// ---------------- model constants ----------------
#define Bsz     256
#define D_Lc    300
#define D_Ac    74
#define DH_Lc   64
#define DH_Ac   16
#define MEMD    128
#define ATT_INc 160
#define GAMMA_INc 288
#define H_ATT1c 256
#define H_ATT2c 64
#define H_Gc    64
#define H_OUTc  64
#define FINAL_INc 208

// K dims padded to multiples of 32 for 16x16x32 bf16 WMMA
#define KXL 320   // D_L 300 -> 320
#define KXA 96    // D_A 74  -> 96
#define KHL 64
#define KHA 32    // 16 -> 32

#define NBLK 16
#define NTHR 256
#define NWAVES (NBLK * (NTHR / 32))
#define ROWS_WG (Bsz / NBLK)      // 16 batch rows staged per workgroup

typedef __attribute__((ext_vector_type(16))) __bf16 v16bf;
typedef __attribute__((ext_vector_type(8)))  float  v8f;

// ---------------- Tensor Data Mover (gfx1250 TDM) ----------------
#if defined(__has_builtin)
#  if __has_builtin(__builtin_amdgcn_tensor_load_to_lds)
#    define HAVE_TDM 1
#  endif
#endif
#ifndef HAVE_TDM
#  define HAVE_TDM 0
#endif

#if HAVE_TDM
typedef unsigned int u32x4 __attribute__((ext_vector_type(4)));
typedef int          i32x4 __attribute__((ext_vector_type(4)));
typedef int          i32x8 __attribute__((ext_vector_type(8)));

// DMA a 2D tile (rows x dim0 fp32 elements, row stride stride0 elements) from
// global memory into LDS at byte offset lds_off. D# layout per ISA 08 §8.3/8.4.
__device__ __forceinline__ void tdm_load_2d(unsigned lds_off, const float* gptr,
                                            unsigned dim0, unsigned rows,
                                            unsigned long long stride0) {
  unsigned long long ga = (unsigned long long)(size_t)gptr;
  u32x4 g0;
  g0[0] = 1u;                                                  // count=1, user mode
  g0[1] = lds_off;                                             // D#.lds_addr
  g0[2] = (unsigned)ga;                                        // global_addr[31:0]
  g0[3] = (unsigned)((ga >> 32) & 0x01ffffffull) | 0x80000000u; // addr[56:32] | type=2
  i32x8 g1;
  g1[0] = 0x00020000;                                          // data_size=4B, mask=0
  g1[1] = (int)((dim0 & 0xffffu) << 16);                       // tensor_dim0 lo16
  g1[2] = (int)((dim0 >> 16) | ((rows & 0xffffu) << 16));      // dim0 hi | tensor_dim1 lo
  g1[3] = (int)((rows >> 16) | ((dim0 & 0xffffu) << 16));      // dim1 hi | tile_dim0
  g1[4] = (int)(rows & 0xffffu);                               // tile_dim1 (tile_dim2=0)
  g1[5] = (int)(unsigned)stride0;                              // tensor_dim0_stride[31:0]
  g1[6] = (int)((stride0 >> 32) & 0xffffull);                  // stride[47:32], dim1_stride=0
  g1[7] = 0;
  i32x4 gz = {0, 0, 0, 0};                                     // groups 2/3: <=2D tensor
#if __clang_major__ >= 23
  i32x8 gz8 = {0, 0, 0, 0, 0, 0, 0, 0};
  __builtin_amdgcn_tensor_load_to_lds(g0, g1, gz, gz, gz8, 0);
#else
  __builtin_amdgcn_tensor_load_to_lds(g0, g1, gz, gz, 0);
#endif
}
#endif  // HAVE_TDM

// ---------------- fragment loaders (ISA 7.12.2 layouts) ----------------
__device__ __forceinline__ v16bf load_frag_a(const __bf16* A, int lda,
                                             int mB, int kB, int l16, int half) {
  const __bf16* p = A + (size_t)(mB + l16) * lda + kB + half * 8;
  union { uint4 q[2]; v16bf v; } u;
  u.q[0] = *(const uint4*)p;
  u.q[1] = *(const uint4*)(p + 16);
  return u.v;
}

__device__ __forceinline__ v16bf load_frag_b(const __bf16* W, int ldw,
                                             int nB, int kB, int l16, int half) {
  const __bf16* p = W + (size_t)(nB + l16) * ldw + kB + half * 8;
  union { uint4 q[2]; v16bf v; } u;
  u.q[0] = *(const uint4*)p;
  u.q[1] = *(const uint4*)(p + 16);
  return u.v;
}

__device__ __forceinline__ float sigf(float x) { return 1.f / (1.f + __expf(-x)); }

__device__ __forceinline__ float actf(float v, int act) {
  if (act == 1) return fmaxf(v, 0.f);
  if (act == 2) return sigf(v);
  if (act == 3) return tanhf(v);
  return v;
}

// Fused GEMM stage: C = act( A1 @ W1^T [+ A2 @ W2^T] + b1 [+ b2] )
__device__ void gemm2(const __bf16* A1, int lda1, int K1,
                      const __bf16* A2, int lda2, int K2,
                      const __bf16* W1, const __bf16* W2,
                      const float* b1, const float* b2,
                      float* Cf, __bf16* Cb, int ldc,
                      int M, int N, int act, int gwave) {
  const int lane = threadIdx.x & 31, l16 = lane & 15, half = lane >> 4;
  const int nT = N >> 4;
  const int tiles = (M >> 4) * nT;
  for (int t = gwave; t < tiles; t += NWAVES) {
    const int mB = (t / nT) << 4, nB = (t % nT) << 4;
    v8f c = {};
    for (int k = 0; k < K1; k += 32) {
      v16bf a = load_frag_a(A1, lda1, mB, k, l16, half);
      v16bf b = load_frag_b(W1, K1, nB, k, l16, half);
      c = __builtin_amdgcn_wmma_f32_16x16x32_bf16(false, a, false, b,
                                                  (short)0, c, false, false);
    }
    if (A2) {
      for (int k = 0; k < K2; k += 32) {
        v16bf a = load_frag_a(A2, lda2, mB, k, l16, half);
        v16bf b = load_frag_b(W2, K2, nB, k, l16, half);
        c = __builtin_amdgcn_wmma_f32_16x16x32_bf16(false, a, false, b,
                                                    (short)0, c, false, false);
      }
    }
    const int col = nB + l16;
    const float bias = (b1 ? b1[col] : 0.f) + (b2 ? b2[col] : 0.f);
#pragma unroll
    for (int r = 0; r < 8; ++r) {
      const int row = mB + r + half * 8;
      float v = actf(c[r] + bias, act);
      if (Cf) Cf[(size_t)row * ldc + col] = v;
      if (Cb) Cb[(size_t)row * ldc + col] = (__bf16)v;
    }
  }
}

// ---------------- grid-wide barrier ----------------
__device__ void grid_sync(unsigned* s) {
  __syncthreads();
  if (threadIdx.x == 0) {
    __threadfence();
    volatile unsigned* vs = (volatile unsigned*)s;
    const unsigned g = vs[1];
    if (atomicAdd(&s[0], 1u) + 1u == (unsigned)NBLK) {
      vs[0] = 0u;
      __threadfence();
      vs[1] = g + 1u;
    } else {
      while (vs[1] == g) __builtin_amdgcn_s_sleep(2);
    }
    __threadfence();
  }
  // NOP when not launched as a workgroup cluster (ISA: ClusterID==0 -> S_NOP);
  // real HW cluster barrier under a cluster launch.
  __builtin_amdgcn_s_cluster_barrier();
  __syncthreads();
}

// ---------------- params ----------------
struct P {
  const float *x_l, *x_a;
  const int* len;
  const float *bihL, *bhhL, *bihA, *bhhA;
  const float *bA1, *bA2, *bA3, *bA4, *bG1a, *bG1b, *bG2a, *bG2b;
  const float *wO1, *bO1, *wO2, *bO2;
  const __bf16 *wXL, *wHL, *wXA, *wHA, *wA1, *wA2, *wA3, *wA4,
               *wG1a, *wG1b, *wG2a, *wG2b;
  __bf16 *Xl, *Xa, *hlb, *hab, *cSb, *z1b, *bothb, *r3b, *rg1b, *rg2b;
  float *gl, *ga, *hl, *cl, *ha, *ca, *mem, *nhl, *ncl, *nha, *nca,
        *cSf, *sf, *cHat, *g1, *g2;
  unsigned* sync;
  float* out;
  int T;
};

// fp32 -> bf16 weight conversion with K padding
__global__ void mfn_convert_pad(const float* __restrict__ src, __bf16* __restrict__ dst,
                                int N, int K, int Kpad) {
  int i = blockIdx.x * 256 + threadIdx.x;
  if (i >= N * Kpad) return;
  int n = i / Kpad, k = i - n * Kpad;
  dst[i] = (__bf16)((k < K) ? src[(size_t)n * K + k] : 0.f);
}

// ---------------- persistent recurrent kernel ----------------
__global__ void __launch_bounds__(NTHR) mfn_persistent(P p) {
  const int gtid = blockIdx.x * NTHR + threadIdx.x;
  const int gwave = gtid >> 5;
  const int stride = NBLK * NTHR;
  const int bRow0 = blockIdx.x * ROWS_WG;

  __shared__ float lds_xl[ROWS_WG * D_Lc];   // 19.2 KB TDM landing pad
  __shared__ float lds_xa[ROWS_WG * D_Ac];   //  4.7 KB

  // init state
  for (int i = gtid; i < Bsz * DH_Lc; i += stride) {
    p.hl[i] = 0.f; p.cl[i] = 0.f; p.hlb[i] = (__bf16)0.f;
  }
  for (int i = gtid; i < Bsz * DH_Ac; i += stride) { p.ha[i] = 0.f; p.ca[i] = 0.f; }
  for (int i = gtid; i < Bsz * KHA; i += stride) p.hab[i] = (__bf16)0.f;
  for (int i = gtid; i < Bsz * MEMD; i += stride) p.mem[i] = 0.f;
  grid_sync(p.sync);

  for (int t = 0; t < p.T; ++t) {
    // --- stage 0: TDM-stage x_l[:,t,:], x_a[:,t,:] tiles into LDS, then bf16 ---
#if HAVE_TDM
    if ((threadIdx.x >> 5) == 0) {   // wave 0 drives the TDM for this WGP
      // flat shared address truncates to the LDS byte offset (ISA 10.2 aperture)
      tdm_load_2d((unsigned)(size_t)&lds_xl[0],
                  p.x_l + (size_t)bRow0 * p.T * D_Lc + (size_t)t * D_Lc,
                  D_Lc, ROWS_WG, (unsigned long long)p.T * D_Lc);
      tdm_load_2d((unsigned)(size_t)&lds_xa[0],
                  p.x_a + (size_t)bRow0 * p.T * D_Ac + (size_t)t * D_Ac,
                  D_Ac, ROWS_WG, (unsigned long long)p.T * D_Ac);
      __builtin_amdgcn_s_wait_tensorcnt(0);
    }
    if (threadIdx.x < ROWS_WG && t + 1 < p.T)   // global_prefetch_b8 of next step
      __builtin_prefetch(p.x_l + (size_t)(bRow0 + threadIdx.x) * p.T * D_Lc +
                         (size_t)(t + 1) * D_Lc, 0, 1);
    __syncthreads();
    for (int i = threadIdx.x; i < ROWS_WG * KXL; i += NTHR) {
      int r = i / KXL, k = i - r * KXL;
      float v = (k < D_Lc) ? lds_xl[r * D_Lc + k] : 0.f;
      p.Xl[(size_t)(bRow0 + r) * KXL + k] = (__bf16)v;
    }
    for (int i = threadIdx.x; i < ROWS_WG * KXA; i += NTHR) {
      int r = i / KXA, k = i - r * KXA;
      float v = (k < D_Ac) ? lds_xa[r * D_Ac + k] : 0.f;
      p.Xa[(size_t)(bRow0 + r) * KXA + k] = (__bf16)v;
    }
#else
    (void)lds_xl; (void)lds_xa;
    for (int i = gtid; i < Bsz * KXL; i += stride) {
      int b = i / KXL, k = i - b * KXL;
      const size_t base = (size_t)b * p.T * D_Lc + (size_t)t * D_Lc;
      float v = (k < D_Lc) ? p.x_l[base + k] : 0.f;
      p.Xl[i] = (__bf16)v;
      if (((k & 63) == 0) && (t + 1 < p.T) && (k < D_Lc))
        __builtin_prefetch(&p.x_l[base + D_Lc + k], 0, 1);
    }
    for (int i = gtid; i < Bsz * KXA; i += stride) {
      int b = i / KXA, k = i - b * KXA;
      float v = (k < D_Ac) ? p.x_a[(size_t)b * p.T * D_Ac + (size_t)t * D_Ac + k] : 0.f;
      p.Xa[i] = (__bf16)v;
    }
#endif
    grid_sync(p.sync);

    // --- stage 1: LSTM gate pre-activations (x@Wih^T + h@Whh^T + bih + bhh) ---
    gemm2(p.Xl, KXL, KXL, p.hlb, KHL, KHL, p.wXL, p.wHL, p.bihL, p.bhhL,
          p.gl, nullptr, 4 * DH_Lc, Bsz, 4 * DH_Lc, 0, gwave);
    gemm2(p.Xa, KXA, KXA, p.hab, KHA, KHA, p.wXA, p.wHA, p.bihA, p.bhhA,
          p.ga, nullptr, 4 * DH_Ac, Bsz, 4 * DH_Ac, 0, gwave);
    grid_sync(p.sync);

    // --- stage 2: LSTM cells (fp32) + build cStar = [c_l, c_a, nc_l, nc_a] ---
    for (int i = gtid; i < Bsz * DH_Lc; i += stride) {
      int b = i >> 6, j = i & 63;
      const float* g = p.gl + (size_t)b * 256;
      float ci = g[j], cf = g[64 + j], cg = g[128 + j], co = g[192 + j];
      float c_old = p.cl[i];
      float cn = sigf(cf) * c_old + sigf(ci) * tanhf(cg);
      float hn = sigf(co) * tanhf(cn);
      p.ncl[i] = cn; p.nhl[i] = hn;
      p.cSf[b * ATT_INc + j] = c_old;      p.cSb[b * ATT_INc + j] = (__bf16)c_old;
      p.cSf[b * ATT_INc + 80 + j] = cn;    p.cSb[b * ATT_INc + 80 + j] = (__bf16)cn;
    }
    for (int i = gtid; i < Bsz * DH_Ac; i += stride) {
      int b = i >> 4, j = i & 15;
      const float* g = p.ga + (size_t)b * 64;
      float ci = g[j], cf = g[16 + j], cg = g[32 + j], co = g[48 + j];
      float c_old = p.ca[i];
      float cn = sigf(cf) * c_old + sigf(ci) * tanhf(cg);
      float hn = sigf(co) * tanhf(cn);
      p.nca[i] = cn; p.nha[i] = hn;
      p.cSf[b * ATT_INc + 64 + j] = c_old;   p.cSb[b * ATT_INc + 64 + j] = (__bf16)c_old;
      p.cSf[b * ATT_INc + 144 + j] = cn;     p.cSb[b * ATT_INc + 144 + j] = (__bf16)cn;
    }
    grid_sync(p.sync);

    // --- stage 3: z1 = relu(cStar @ A1^T + b1) ---
    gemm2(p.cSb, ATT_INc, ATT_INc, nullptr, 0, 0, p.wA1, nullptr, p.bA1, nullptr,
          nullptr, p.z1b, H_ATT1c, Bsz, H_ATT1c, 1, gwave);
    grid_sync(p.sync);

    // --- stage 4: s = z1 @ A2^T + b2 ---
    gemm2(p.z1b, H_ATT1c, H_ATT1c, nullptr, 0, 0, p.wA2, nullptr, p.bA2, nullptr,
          p.sf, nullptr, ATT_INc, Bsz, ATT_INc, 0, gwave);
    grid_sync(p.sync);

    // --- stage 5: row softmax, attended = att*cStar, both = [attended, mem] ---
    if (gtid < Bsz) {
      float* s = p.sf + gtid * ATT_INc;
      float m = s[0];
      for (int k = 1; k < ATT_INc; ++k) m = fmaxf(m, s[k]);
      float sum = 0.f;
      for (int k = 0; k < ATT_INc; ++k) { float e = __expf(s[k] - m); s[k] = e; sum += e; }
      float inv = 1.f / sum;
      __bf16* bo = p.bothb + gtid * GAMMA_INc;
      for (int k = 0; k < ATT_INc; ++k)
        bo[k] = (__bf16)(s[k] * inv * p.cSf[gtid * ATT_INc + k]);
      for (int k = 0; k < MEMD; ++k)
        bo[ATT_INc + k] = (__bf16)p.mem[gtid * MEMD + k];
    }
    grid_sync(p.sync);

    // --- stage 6 (independent): r3, rg1, rg2 hidden layers ---
    gemm2(p.bothb, GAMMA_INc, ATT_INc, nullptr, 0, 0, p.wA3, nullptr, p.bA3, nullptr,
          nullptr, p.r3b, H_ATT2c, Bsz, H_ATT2c, 1, gwave);
    gemm2(p.bothb, GAMMA_INc, GAMMA_INc, nullptr, 0, 0, p.wG1a, nullptr, p.bG1a, nullptr,
          nullptr, p.rg1b, H_Gc, Bsz, H_Gc, 1, gwave);
    gemm2(p.bothb, GAMMA_INc, GAMMA_INc, nullptr, 0, 0, p.wG2a, nullptr, p.bG2a, nullptr,
          nullptr, p.rg2b, H_Gc, Bsz, H_Gc, 1, gwave);
    grid_sync(p.sync);

    // --- stage 7 (independent): cHat = tanh(...), g1/g2 = sigmoid(...) ---
    gemm2(p.r3b, H_ATT2c, H_ATT2c, nullptr, 0, 0, p.wA4, nullptr, p.bA4, nullptr,
          p.cHat, nullptr, MEMD, Bsz, MEMD, 3, gwave);
    gemm2(p.rg1b, H_Gc, H_Gc, nullptr, 0, 0, p.wG1b, nullptr, p.bG1b, nullptr,
          p.g1, nullptr, MEMD, Bsz, MEMD, 2, gwave);
    gemm2(p.rg2b, H_Gc, H_Gc, nullptr, 0, 0, p.wG2b, nullptr, p.bG2b, nullptr,
          p.g2, nullptr, MEMD, Bsz, MEMD, 2, gwave);
    grid_sync(p.sync);

    // --- stage 8: masked state commit + bf16 h staging for next step ---
    for (int i = gtid; i < Bsz * MEMD; i += stride) {
      int b = i >> 7;
      if (t < p.len[b]) p.mem[i] = p.g1[i] * p.mem[i] + p.g2[i] * p.cHat[i];
    }
    for (int i = gtid; i < Bsz * DH_Lc; i += stride) {
      int b = i >> 6;
      if (t < p.len[b]) { p.hl[i] = p.nhl[i]; p.cl[i] = p.ncl[i]; }
      p.hlb[i] = (__bf16)p.hl[i];
    }
    for (int i = gtid; i < Bsz * DH_Ac; i += stride) {
      int b = i >> 4, j = i & 15;
      if (t < p.len[b]) { p.ha[i] = p.nha[i]; p.ca[i] = p.nca[i]; }
      p.hab[b * KHA + j] = (__bf16)p.ha[i];
      p.hab[b * KHA + 16 + j] = (__bf16)0.f;   // zero pad K 16->32
    }
    grid_sync(p.sync);
  }

  // --- output head: out = relu([h_l,h_a,mem] @ W1^T + b1) @ w2^T + b2 ---
  if (blockIdx.x == 0 && threadIdx.x < Bsz) {
    const int b = threadIdx.x;
    float acc = p.bO2[0];
    for (int j = 0; j < H_OUTc; ++j) {
      const float* w = p.wO1 + (size_t)j * FINAL_INc;
      float h = p.bO1[j];
      for (int k = 0; k < DH_Lc; ++k) h += w[k] * p.hl[b * DH_Lc + k];
      for (int k = 0; k < DH_Ac; ++k) h += w[64 + k] * p.ha[b * DH_Ac + k];
      for (int k = 0; k < MEMD; ++k) h += w[80 + k] * p.mem[b * MEMD + k];
      acc += p.wO2[j] * fmaxf(h, 0.f);
    }
    p.out[b] = acc;
  }
}

// ---------------- host ----------------
extern "C" void kernel_launch(void* const* d_in, const int* in_sizes, int n_in,
                              void* d_out, int out_size, void* d_ws, size_t ws_size,
                              hipStream_t stream) {
  P p{};
  p.x_l = (const float*)d_in[0];
  p.x_a = (const float*)d_in[1];
  p.len = (const int*)d_in[2];
  const float* fWihL = (const float*)d_in[3];
  const float* fWhhL = (const float*)d_in[4];
  p.bihL = (const float*)d_in[5];
  p.bhhL = (const float*)d_in[6];
  const float* fWihA = (const float*)d_in[7];
  const float* fWhhA = (const float*)d_in[8];
  p.bihA = (const float*)d_in[9];
  p.bhhA = (const float*)d_in[10];
  const float* fA1 = (const float*)d_in[11];  p.bA1 = (const float*)d_in[12];
  const float* fA2 = (const float*)d_in[13];  p.bA2 = (const float*)d_in[14];
  const float* fA3 = (const float*)d_in[15];  p.bA3 = (const float*)d_in[16];
  const float* fA4 = (const float*)d_in[17];  p.bA4 = (const float*)d_in[18];
  const float* fG1a = (const float*)d_in[19]; p.bG1a = (const float*)d_in[20];
  const float* fG1b = (const float*)d_in[21]; p.bG1b = (const float*)d_in[22];
  const float* fG2a = (const float*)d_in[23]; p.bG2a = (const float*)d_in[24];
  const float* fG2b = (const float*)d_in[25]; p.bG2b = (const float*)d_in[26];
  p.wO1 = (const float*)d_in[27];  p.bO1 = (const float*)d_in[28];
  p.wO2 = (const float*)d_in[29];  p.bO2 = (const float*)d_in[30];
  p.T = in_sizes[0] / (Bsz * D_Lc);
  p.out = (float*)d_out;

  char* ws = (char*)d_ws;
  size_t off = 0;
  auto alloc = [&](size_t bytes) -> void* {
    void* q = ws + off;
    off += (bytes + 255) & ~(size_t)255;
    return q;
  };

  p.sync = (unsigned*)alloc(256);
  hipMemsetAsync(p.sync, 0, 256, stream);

  auto cw = [&](const float* src, int N, int K, int Kpad) -> __bf16* {
    __bf16* dst = (__bf16*)alloc((size_t)N * Kpad * sizeof(__bf16));
    int n = N * Kpad;
    mfn_convert_pad<<<(n + 255) / 256, 256, 0, stream>>>(src, dst, N, K, Kpad);
    return dst;
  };
  p.wXL  = cw(fWihL, 4 * DH_Lc, D_Lc, KXL);
  p.wHL  = cw(fWhhL, 4 * DH_Lc, DH_Lc, KHL);
  p.wXA  = cw(fWihA, 4 * DH_Ac, D_Ac, KXA);
  p.wHA  = cw(fWhhA, 4 * DH_Ac, DH_Ac, KHA);
  p.wA1  = cw(fA1,  H_ATT1c, ATT_INc, ATT_INc);
  p.wA2  = cw(fA2,  ATT_INc, H_ATT1c, H_ATT1c);
  p.wA3  = cw(fA3,  H_ATT2c, ATT_INc, ATT_INc);
  p.wA4  = cw(fA4,  MEMD,    H_ATT2c, H_ATT2c);
  p.wG1a = cw(fG1a, H_Gc,    GAMMA_INc, GAMMA_INc);
  p.wG1b = cw(fG1b, MEMD,    H_Gc, H_Gc);
  p.wG2a = cw(fG2a, H_Gc,    GAMMA_INc, GAMMA_INc);
  p.wG2b = cw(fG2b, MEMD,    H_Gc, H_Gc);

  p.Xl   = (__bf16*)alloc((size_t)Bsz * KXL * 2);
  p.Xa   = (__bf16*)alloc((size_t)Bsz * KXA * 2);
  p.hlb  = (__bf16*)alloc((size_t)Bsz * KHL * 2);
  p.hab  = (__bf16*)alloc((size_t)Bsz * KHA * 2);
  p.cSb  = (__bf16*)alloc((size_t)Bsz * ATT_INc * 2);
  p.z1b  = (__bf16*)alloc((size_t)Bsz * H_ATT1c * 2);
  p.bothb = (__bf16*)alloc((size_t)Bsz * GAMMA_INc * 2);
  p.r3b  = (__bf16*)alloc((size_t)Bsz * H_ATT2c * 2);
  p.rg1b = (__bf16*)alloc((size_t)Bsz * H_Gc * 2);
  p.rg2b = (__bf16*)alloc((size_t)Bsz * H_Gc * 2);

  p.gl   = (float*)alloc((size_t)Bsz * 256 * 4);
  p.ga   = (float*)alloc((size_t)Bsz * 64 * 4);
  p.hl   = (float*)alloc((size_t)Bsz * DH_Lc * 4);
  p.cl   = (float*)alloc((size_t)Bsz * DH_Lc * 4);
  p.ha   = (float*)alloc((size_t)Bsz * DH_Ac * 4);
  p.ca   = (float*)alloc((size_t)Bsz * DH_Ac * 4);
  p.mem  = (float*)alloc((size_t)Bsz * MEMD * 4);
  p.nhl  = (float*)alloc((size_t)Bsz * DH_Lc * 4);
  p.ncl  = (float*)alloc((size_t)Bsz * DH_Lc * 4);
  p.nha  = (float*)alloc((size_t)Bsz * DH_Ac * 4);
  p.nca  = (float*)alloc((size_t)Bsz * DH_Ac * 4);
  p.cSf  = (float*)alloc((size_t)Bsz * ATT_INc * 4);
  p.sf   = (float*)alloc((size_t)Bsz * ATT_INc * 4);
  p.cHat = (float*)alloc((size_t)Bsz * MEMD * 4);
  p.g1   = (float*)alloc((size_t)Bsz * MEMD * 4);
  p.g2   = (float*)alloc((size_t)Bsz * MEMD * 4);

  mfn_persistent<<<NBLK, NTHR, 0, stream>>>(p);
}
